// EntityEncoder_30777735643591
// MI455X (gfx1250) — matrix operand order repaired and checked by reading, not compiled, not run on hardware
//
#include <hip/hip_runtime.h>
#include <hip/hip_bf16.h>
#include <math.h>

#define D_DIM 768

typedef __attribute__((ext_vector_type(16))) __bf16 v16bf;
typedef __attribute__((ext_vector_type(8)))  float  v8f;

// ---------- helpers ----------

__device__ __forceinline__ unsigned short f2bf(float f) {
    unsigned int u = __float_as_uint(f);
    unsigned int r = u + 0x7FFFu + ((u >> 16) & 1u);   // round-to-nearest-even
    return (unsigned short)(r >> 16);
}

__device__ __forceinline__ float wave_sum(float v) {
    #pragma unroll
    for (int off = 16; off > 0; off >>= 1) v += __shfl_down(v, off, 32);
    return v;
}

__device__ __forceinline__ float wave_max(float v) {
    #pragma unroll
    for (int off = 16; off > 0; off >>= 1) v = fmaxf(v, __shfl_down(v, off, 32));
    return v;
}

__device__ __forceinline__ v16bf ld16bf(const unsigned short* p0, const unsigned short* p1) {
    union { v16bf v; uint4 q[2]; } u;
    u.q[0] = *reinterpret_cast<const uint4*>(p0);
    u.q[1] = *reinterpret_cast<const uint4*>(p1);
    return u.v;
}

// CDNA5 async global->LDS copy (ASYNCcnt-tracked). vdst = LDS byte address,
// vaddr = 64-bit global address (GV mode, SADDR=off).
__device__ __forceinline__ void async_g2l_b128(unsigned lds_byte_addr, const void* gaddr) {
    asm volatile("global_load_async_to_lds_b128 %0, %1, off"
                 :: "v"(lds_byte_addr), "v"(gaddr) : "memory");
}

__device__ __forceinline__ void wait_asynccnt0() {
#if __has_builtin(__builtin_amdgcn_s_wait_asynccnt)
    __builtin_amdgcn_s_wait_asynccnt(0);
#else
    asm volatile("s_wait_asynccnt 0x0" ::: "memory");
#endif
}

// ---------- phase 0: f32 -> bf16 conversion (weights) ----------

__global__ void k_f32_to_bf16(const float* __restrict__ in, unsigned short* __restrict__ out, long n) {
    long i = (long)blockIdx.x * blockDim.x + threadIdx.x;
    long stride = (long)gridDim.x * blockDim.x;
    for (; i < n; i += stride) out[i] = f2bf(in[i]);
}

// ---------- phase 1: score LUTs (prompt / count) ----------

__global__ void k_row_dots(const float* __restrict__ rows, const float* __restrict__ w,
                           float* __restrict__ out, int nrows) {
    int wid  = (blockIdx.x * blockDim.x + threadIdx.x) >> 5;
    int lane = threadIdx.x & 31;
    if (wid >= nrows) return;
    const float* r = rows + (size_t)wid * D_DIM;
    float acc = 0.f;
    for (int k = lane; k < D_DIM; k += 32) acc += r[k] * w[k];
    acc = wave_sum(acc);
    if (lane == 0) out[wid] = acc;
}

// ---------- phase 2: per-edge score ----------

__global__ void k_edge_score(const float* __restrict__ ent, const float* __restrict__ nb,
                             const float* __restrict__ rel, const float* __restrict__ sW,
                             const float* __restrict__ sb,
                             const float* __restrict__ pS, const float* __restrict__ cS,
                             const int* __restrict__ counts, const int* __restrict__ pidx,
                             float* __restrict__ score, int N) {
    int wid  = (blockIdx.x * blockDim.x + threadIdx.x) >> 5;
    int lane = threadIdx.x & 31;
    if (wid >= N) return;
    const float* w1 = sW + D_DIM;
    const float* w2 = sW + 2 * D_DIM;
    const float* w3 = sW + 3 * D_DIM;
    size_t base = (size_t)wid * D_DIM;
    float acc = 0.f;
    for (int k = lane; k < D_DIM; k += 32)
        acc += ent[base + k] * w1[k] + nb[base + k] * w2[k] + rel[base + k] * w3[k];
    acc = wave_sum(acc);
    if (lane == 0)
        score[wid] = acc + pS[pidx[wid]] + cS[counts[wid]] + sb[0];
}

// ---------- phase 3: segment offsets from sorted indices ----------

__global__ void k_seg_offsets(const int* __restrict__ eidx, int* __restrict__ segoff, int N, int E) {
    int i = blockIdx.x * blockDim.x + threadIdx.x;
    if (i >= N) return;
    int cur  = eidx[i];
    int prev = (i == 0) ? -1 : eidx[i - 1];
    for (int e = prev + 1; e <= cur; ++e) segoff[e] = i;
    if (i == N - 1)
        for (int e = cur + 1; e <= E; ++e) segoff[e] = N;
}

// ---------- phase 4: segment max + sum(exp) ----------

__global__ void k_seg_stats(const float* __restrict__ score, const int* __restrict__ segoff,
                            float* __restrict__ segM, float* __restrict__ segS, int E) {
    int wid  = (blockIdx.x * blockDim.x + threadIdx.x) >> 5;
    int lane = threadIdx.x & 31;
    if (wid >= E) return;
    int beg = segoff[wid], end = segoff[wid + 1];
    if (end <= beg) {
        if (lane == 0) { segM[wid] = 0.f; segS[wid] = 1.f; }
        return;
    }
    float m = -INFINITY;
    for (int j = beg + lane; j < end; j += 32) m = fmaxf(m, score[j]);
    m = wave_max(m);
    m = __shfl(m, 0, 32);
    float s = 0.f;
    for (int j = beg + lane; j < end; j += 32) s += __expf(score[j] - m);
    s = wave_sum(s);
    if (lane == 0) { segM[wid] = m; segS[wid] = s; }
}

// ---------- phase 5: edge attention (in place) ----------

__global__ void k_edge_attn(float* __restrict__ score, const int* __restrict__ eidx,
                            const float* __restrict__ segM, const float* __restrict__ segS, int N) {
    int i = blockIdx.x * blockDim.x + threadIdx.x;
    if (i >= N) return;
    int e = eidx[i];
    score[i] = __expf(score[i] - segM[e]) / segS[e];
}

// ---------- phase 6: segment aggregation -> bf16 aggregates ----------

__global__ void k_seg_aggregate(const float* __restrict__ attn, const int* __restrict__ segoff,
                                const float* __restrict__ rel, const float* __restrict__ ctab,
                                const int* __restrict__ counts, const float* __restrict__ ent,
                                unsigned short* __restrict__ relAgg, unsigned short* __restrict__ entAgg,
                                int E) {
    int wid  = (blockIdx.x * blockDim.x + threadIdx.x) >> 5;
    int lane = threadIdx.x & 31;
    if (wid >= E) return;
    int beg = segoff[wid], end = segoff[wid + 1];
    int cnt = end - beg;
    unsigned short* ra = relAgg + (size_t)wid * (2 * D_DIM);
    unsigned short* ea = entAgg + (size_t)wid * D_DIM;
    if (cnt == 0) {
        for (int d = lane; d < 2 * D_DIM; d += 32) ra[d] = 0;
        for (int d = lane; d < D_DIM; d += 32) ea[d] = 0;
        return;
    }
    float inv = 1.0f / (float)cnt;
    for (int d = lane; d < D_DIM; d += 32) {
        float aR = 0.f, aC = 0.f, aE = 0.f;
        for (int j = beg; j < end; ++j) {
            float w = attn[j];
            aR += w * rel[(size_t)j * D_DIM + d];
            aC += w * ctab[(size_t)counts[j] * D_DIM + d];
            aE += ent[(size_t)j * D_DIM + d];
        }
        ra[d]         = f2bf(aR);
        ra[D_DIM + d] = f2bf(aC);
        ea[d]         = f2bf(aE * inv);
    }
}

// ---------- phase 7: bf16 WMMA GEMM with async-LDS staged B ----------
// Block = 8 waves covering 16 M-tiles x one shared 64-wide N tile.
// B tile (64x32 bf16) staged in LDS via global_load_async_to_lds_b128,
// double-buffered, prefetch clamped (no per-iteration branch). Each wave owns
// two 16-row M tiles -> 8 WMMAs per K-step between barriers.
// out[M,Nmat] = A[M,K] * W[Nmat,K]^T + bias.
// LDS layout: Bsm[buf][col 0..63][k 0..31] with 80B row stride (bank spread).

#define BSTRIDE 40  // ushorts per LDS row (80 bytes)

__global__ void __launch_bounds__(256)
k_gemm_bf16(const unsigned short* __restrict__ A, const unsigned short* __restrict__ B,
            const float* __restrict__ bias, float* __restrict__ out,
            int M, int Nmat, int K) {
    __shared__ __align__(16) unsigned short Bsm[2][64][BSTRIDE];

    const int tid    = threadIdx.x;
    const int lane   = tid & 31;
    const int wave   = tid >> 5;
    const int half   = lane >> 4;
    const int l16    = lane & 15;
    const int tilesN = Nmat / 64;
    const int tilesM = M / 16;

    const int bm = blockIdx.x / tilesN;
    const int bn = blockIdx.x % tilesN;
    int tm0 = bm * 16 + wave;
    int tm1 = bm * 16 + wave + 8;
    const bool st0 = (tm0 < tilesM);
    const bool st1 = (tm1 < tilesM);
    if (tm0 >= tilesM) tm0 = tilesM - 1;        // clamp: duplicate compute, skip store
    if (tm1 >= tilesM) tm1 = tilesM - 1;
    const int nbase0 = bn * 64;

    // this thread's async-copy slot: one 16B chunk of the 64x32 B tile
    const int fcol = tid >> 2;                  // 0..63
    const int fseg = tid & 3;                   // 0..3 (8 bf16 each)
    const unsigned short* gB = B + (size_t)(nbase0 + fcol) * K + fseg * 8;
    const unsigned ldsBase = (unsigned)(unsigned long long)(const void*)&Bsm[0][0][0];
    const unsigned myLds   = ldsBase + (unsigned)(fcol * (BSTRIDE * 2) + fseg * 16);
    const unsigned bufStep = 64u * BSTRIDE * 2u;

    const unsigned short* Ap0 = A + (size_t)(tm0 * 16 + l16) * K;
    const unsigned short* Ap1 = A + (size_t)(tm1 * 16 + l16) * K;

    // prologue: stage k0 = 0 into buffer 0
    async_g2l_b128(myLds, gB);

    v8f c00 = {}, c01 = {}, c02 = {}, c03 = {};
    v8f c10 = {}, c11 = {}, c12 = {}, c13 = {};
    int s = 0;
    for (int k0 = 0; k0 < K; k0 += 32, s ^= 1) {
        wait_asynccnt0();          // my async writes for buf[s] landed in LDS
        __syncthreads();           // everyone's writes landed; prior reads of buf[s^1] done

        // branch-free prefetch: clamp so last iteration re-fetches the tail tile
        int knext = k0 + 32;
        if (knext > K - 32) knext = K - 32;
        async_g2l_b128(myLds + (unsigned)(s ^ 1) * bufStep, gB + knext);

        // A fragments straight from global (hot in L2)
        const int ka = k0 + 16 * half;
        v16bf a0 = ld16bf(Ap0 + ka, Ap0 + ka + 8);
        v16bf a1 = ld16bf(Ap1 + ka, Ap1 + ka + 8);

        // B fragments from LDS: column = subtile*16 + l16, 16 contiguous K at 16*half
        const unsigned short* r0 = &Bsm[s][ 0 + l16][half * 16];
        const unsigned short* r1 = &Bsm[s][16 + l16][half * 16];
        const unsigned short* r2 = &Bsm[s][32 + l16][half * 16];
        const unsigned short* r3 = &Bsm[s][48 + l16][half * 16];
        v16bf b0 = ld16bf(r0, r0 + 8);
        v16bf b1 = ld16bf(r1, r1 + 8);
        v16bf b2 = ld16bf(r2, r2 + 8);
        v16bf b3 = ld16bf(r3, r3 + 8);

        c00 = __builtin_amdgcn_wmma_f32_16x16x32_bf16(false, a0, false, b0, (short)0, c00, false, false);
        c10 = __builtin_amdgcn_wmma_f32_16x16x32_bf16(false, a1, false, b0, (short)0, c10, false, false);
        c01 = __builtin_amdgcn_wmma_f32_16x16x32_bf16(false, a0, false, b1, (short)0, c01, false, false);
        c11 = __builtin_amdgcn_wmma_f32_16x16x32_bf16(false, a1, false, b1, (short)0, c11, false, false);
        c02 = __builtin_amdgcn_wmma_f32_16x16x32_bf16(false, a0, false, b2, (short)0, c02, false, false);
        c12 = __builtin_amdgcn_wmma_f32_16x16x32_bf16(false, a1, false, b2, (short)0, c12, false, false);
        c03 = __builtin_amdgcn_wmma_f32_16x16x32_bf16(false, a0, false, b3, (short)0, c03, false, false);
        c13 = __builtin_amdgcn_wmma_f32_16x16x32_bf16(false, a1, false, b3, (short)0, c13, false, false);
    }

    const float bv0 = bias[nbase0 + l16];
    const float bv1 = bias[nbase0 + 16 + l16];
    const float bv2 = bias[nbase0 + 32 + l16];
    const float bv3 = bias[nbase0 + 48 + l16];
    if (st0) {
        #pragma unroll
        for (int r = 0; r < 8; ++r) {
            int m = tm0 * 16 + half * 8 + r;    // 32-bit C/D layout: VGPR r, halves split M by 8
            float* orow = out + (size_t)m * Nmat + nbase0 + l16;
            orow[ 0] = c00[r] + bv0;
            orow[16] = c01[r] + bv1;
            orow[32] = c02[r] + bv2;
            orow[48] = c03[r] + bv3;
        }
    }
    if (st1) {
        #pragma unroll
        for (int r = 0; r < 8; ++r) {
            int m = tm1 * 16 + half * 8 + r;
            float* orow = out + (size_t)m * Nmat + nbase0 + l16;
            orow[ 0] = c10[r] + bv0;
            orow[16] = c11[r] + bv1;
            orow[32] = c12[r] + bv2;
            orow[48] = c13[r] + bv3;
        }
    }
}

// ---------- launcher ----------

extern "C" void kernel_launch(void* const* d_in, const int* in_sizes, int n_in,
                              void* d_out, int out_size, void* d_ws, size_t ws_size,
                              hipStream_t stream) {
    const float* prompt = (const float*)d_in[0];
    const float* entity = (const float*)d_in[1];
    const float* neigh  = (const float*)d_in[2];
    const float* rel    = (const float*)d_in[3];
    const float* ctab   = (const float*)d_in[4];
    const float* sW     = (const float*)d_in[5];
    const float* sb     = (const float*)d_in[6];
    const float* relW   = (const float*)d_in[7];
    const float* relb   = (const float*)d_in[8];
    const float* entW   = (const float*)d_in[9];
    const float* entb   = (const float*)d_in[10];
    const int*   counts = (const int*)d_in[11];
    const int*   pidx   = (const int*)d_in[12];
    const int*   eidx   = (const int*)d_in[13];

    const int D   = D_DIM;
    const int N   = in_sizes[11];
    const int P   = in_sizes[0] / D;
    const int C   = in_sizes[4] / D;
    const int OUT = in_sizes[8];
    const int E   = out_size / (2 * OUT);

    float* rel_out = (float*)d_out;
    float* ent_out = rel_out + (size_t)E * OUT;

    // workspace layout (256B aligned slices)
    char* ws = (char*)d_ws;
    auto take = [&](size_t bytes) { char* p = ws; ws += (bytes + 255) & ~(size_t)255; return p; };
    float*          pS       = (float*)take((size_t)P * 4);
    float*          cS       = (float*)take((size_t)C * 4);
    float*          score    = (float*)take((size_t)N * 4);
    int*            segoff   = (int*)take((size_t)(E + 1) * 4);
    float*          segM     = (float*)take((size_t)E * 4);
    float*          segS     = (float*)take((size_t)E * 4);
    unsigned short* relW16   = (unsigned short*)take((size_t)OUT * 2 * D * 2);
    unsigned short* entW16   = (unsigned short*)take((size_t)OUT * D * 2);
    unsigned short* relAgg16 = (unsigned short*)take((size_t)E * 2 * D * 2);
    unsigned short* entAgg16 = (unsigned short*)take((size_t)E * D * 2);

    // weights -> bf16
    {
        long n1 = (long)OUT * 2 * D;
        long n2 = (long)OUT * D;
        k_f32_to_bf16<<<dim3((unsigned)((n1 + 2047) / 2048)), dim3(256), 0, stream>>>(relW, relW16, n1);
        k_f32_to_bf16<<<dim3((unsigned)((n2 + 2047) / 2048)), dim3(256), 0, stream>>>(entW, entW16, n2);
    }

    // score LUTs
    k_row_dots<<<dim3((P * 32 + 255) / 256), dim3(256), 0, stream>>>(prompt, sW, pS, P);
    k_row_dots<<<dim3((C * 32 + 255) / 256), dim3(256), 0, stream>>>(ctab, sW + 4 * D, cS, C);

    // per-edge score
    k_edge_score<<<dim3((N * 32 + 255) / 256), dim3(256), 0, stream>>>(
        entity, neigh, rel, sW, sb, pS, cS, counts, pidx, score, N);

    // segment offsets
    k_seg_offsets<<<dim3((N + 255) / 256), dim3(256), 0, stream>>>(eidx, segoff, N, E);

    // segment softmax stats
    k_seg_stats<<<dim3((E * 32 + 255) / 256), dim3(256), 0, stream>>>(score, segoff, segM, segS, E);

    // attention in place
    k_edge_attn<<<dim3((N + 255) / 256), dim3(256), 0, stream>>>(score, eidx, segM, segS, N);

    // segment aggregation -> bf16 aggregates
    k_seg_aggregate<<<dim3((E * 32 + 255) / 256), dim3(256), 0, stream>>>(
        score, segoff, rel, ctab, counts, entity, relAgg16, entAgg16, E);

    // projector GEMMs (bf16 WMMA, f32 accumulate, async-LDS staged B)
    {
        int tilesM = E / 16;
        int tilesN = OUT / 64;
        int gridM  = (tilesM + 15) / 16;
        k_gemm_bf16<<<dim3(gridM * tilesN), dim3(256), 0, stream>>>(
            relAgg16, relW16, relb, rel_out, E, OUT, 2 * D);
        k_gemm_bf16<<<dim3(gridM * tilesN), dim3(256), 0, stream>>>(
            entAgg16, entW16, entb, ent_out, E, OUT, D);
    }
}
